// IndexerRef_65197603553723
// MI455X (gfx1250) — 8-bit matrix operand order repaired and checked by reading, compiled, not run on hardware
//
#include <hip/hip_runtime.h>


typedef __attribute__((ext_vector_type(8)))  float v8f;
typedef __attribute__((ext_vector_type(16))) int   v16i;

#define D_DIM   512
#define GROUP   128
#define NGROUP  4
#define SEQ_LEN 8192
#define TOPK    2048
#define FP8_MAX 448.0f

__device__ __forceinline__ float bf2f(unsigned short h) {
    return __uint_as_float(((unsigned int)h) << 16);
}

#if __has_builtin(__builtin_amdgcn_cvt_pk_fp8_f32)
__device__ __forceinline__ int pack4_e4m3(float f0, float f1, float f2, float f3) {
    int w = 0;
    w = __builtin_amdgcn_cvt_pk_fp8_f32(f0, f1, w, false);
    w = __builtin_amdgcn_cvt_pk_fp8_f32(f2, f3, w, true);
    return w;
}
#else
__device__ __forceinline__ unsigned int sw_e4m3(float x) {
    float c = fminf(fmaxf(x, -FP8_MAX), FP8_MAX);
    unsigned int b = __float_as_uint(c);
    unsigned int sign = (b >> 24) & 0x80u;
    unsigned int a = b & 0x7fffffffu;
    if (a < 0x3C800000u) {
        unsigned int q = (unsigned int)__float2int_rn(fabsf(c) * 512.0f);
        return (q > 7u) ? (sign | 0x08u) : (sign | q);
    }
    unsigned int mant = a & 0x7fffffu;
    unsigned int keep = mant >> 20;
    unsigned int rem  = mant & 0xfffffu;
    keep += (rem > 0x80000u || (rem == 0x80000u && (keep & 1u))) ? 1u : 0u;
    int e = (int)(a >> 23) - 127 + 7;
    if (keep == 8u) { keep = 0u; e += 1; }
    if (e > 15 || (e == 15 && keep == 7u)) return sign | 0x7Eu;
    return sign | ((unsigned int)e << 3) | keep;
}
__device__ __forceinline__ int pack4_e4m3(float f0, float f1, float f2, float f3) {
    return (int)(sw_e4m3(f0) | (sw_e4m3(f1) << 8) |
                 (sw_e4m3(f2) << 16) | (sw_e4m3(f3) << 24));
}
#endif

__global__ __launch_bounds__(128) void quant_q_kernel(
    const unsigned short* __restrict__ q,
    unsigned int* __restrict__ q8,
    float* __restrict__ qs)
{
    const int b    = blockIdx.x;
    const int tid  = threadIdx.x;
    const int wave = tid >> 5;
    const int lane = tid & 31;

    ushort4 h = *(const ushort4*)(q + (size_t)b * D_DIM + tid * 4);
    float f0 = bf2f(h.x), f1 = bf2f(h.y), f2 = bf2f(h.z), f3 = bf2f(h.w);
    float a = fmaxf(fmaxf(fabsf(f0), fabsf(f1)), fmaxf(fabsf(f2), fabsf(f3)));
    #pragma unroll
    for (int off = 16; off > 0; off >>= 1)
        a = fmaxf(a, __shfl_xor(a, off, 32));
    a = fmaxf(a, 1e-4f);
    float s  = a * (1.0f / FP8_MAX);
    float rs = FP8_MAX / a;
    q8[(size_t)b * (D_DIM / 4) + tid] =
        (unsigned int)pack4_e4m3(f0 * rs, f1 * rs, f2 * rs, f3 * rs);
    if (lane == 0) qs[b * NGROUP + wave] = s;
}

__global__ __launch_bounds__(256) void score_kernel(
    const unsigned short* __restrict__ K,
    const unsigned char* __restrict__ q8,
    const float* __restrict__ qs,
    float* __restrict__ score,
    int S)
{
    const int lane  = threadIdx.x & 31;
    const int wave  = threadIdx.x >> 5;
    const int blocksPerB = S / 128;
    const int b     = blockIdx.x / blocksPerB;
    const int tile  = (blockIdx.x % blocksPerB) * 128 + wave * 16;
    const int half  = lane >> 4;
    const int m     = lane & 15;

    const unsigned short* krow = K + ((size_t)b * S + (tile + m)) * D_DIM;
    const unsigned char*  qrow = q8 + (size_t)b * D_DIM;
    const float4 qsv = *(const float4*)(qs + b * NGROUP);

    v8f acc = {0.f, 0.f, 0.f, 0.f, 0.f, 0.f, 0.f, 0.f};
    const v8f zeroC = {0.f, 0.f, 0.f, 0.f, 0.f, 0.f, 0.f, 0.f};

    #pragma unroll
    for (int g = 0; g < NGROUP; ++g) {
        ushort4 raw[16];
        float amax = 0.f;
        #pragma unroll
        for (int v = 0; v < 16; ++v) {
            const int kst = (v >> 2) * 32 + half * 16 + (v & 3) * 4;
            raw[v] = *(const ushort4*)(krow + g * GROUP + kst);
            amax = fmaxf(amax, fabsf(bf2f(raw[v].x)));
            amax = fmaxf(amax, fabsf(bf2f(raw[v].y)));
            amax = fmaxf(amax, fabsf(bf2f(raw[v].z)));
            amax = fmaxf(amax, fabsf(bf2f(raw[v].w)));
        }
        amax = fmaxf(amax, __shfl_xor(amax, 16, 32));
        amax = fmaxf(amax, 1e-4f);
        const float ks_scale = amax * (1.0f / FP8_MAX);
        const float rs       = FP8_MAX / amax;

        v16i A;
        #pragma unroll
        for (int v = 0; v < 16; ++v) {
            A[v] = pack4_e4m3(bf2f(raw[v].x) * rs,
                              bf2f(raw[v].y) * rs,
                              bf2f(raw[v].z) * rs,
                              bf2f(raw[v].w) * rs);
        }

        v16i Bq;
        #pragma unroll
        for (int c = 0; c < 4; ++c) {
            int4 t = *(const int4*)(qrow + g * GROUP + c * 32 + half * 16);
            Bq[4 * c + 0] = t.x; Bq[4 * c + 1] = t.y;
            Bq[4 * c + 2] = t.z; Bq[4 * c + 3] = t.w;
        }

        v8f d = __builtin_amdgcn_wmma_f32_16x16x128_fp8_fp8(
            A, Bq, (short)0, zeroC, false, false);

        const float qs_g = (g == 0) ? qsv.x : (g == 1) ? qsv.y : (g == 2) ? qsv.z : qsv.w;
        #pragma unroll
        for (int r = 0; r < 8; ++r) {
            float sm = __shfl(ks_scale, r + 8 * half, 32);
            acc[r] += d[r] * (qs_g * sm);
        }
    }

    if (m == 0) {
        float4 lo = {acc[0], acc[1], acc[2], acc[3]};
        float4 hi = {acc[4], acc[5], acc[6], acc[7]};
        float* out = score + (size_t)b * S + tile + half * 8;
        *(float4*)(out)     = lo;
        *(float4*)(out + 4) = hi;
    }
}

__global__ __launch_bounds__(1024) void topk_kernel(
    const float* __restrict__ score,
    const int* __restrict__ src_page_table,
    const int* __restrict__ row_starts,
    int* __restrict__ pages_out,
    float* __restrict__ vals_out,
    int S)
{
    __shared__ unsigned int   skey[SEQ_LEN];
    __shared__ unsigned short sidx[SEQ_LEN];

    const int b   = blockIdx.x;
    const int tid = threadIdx.x;
    const int st  = row_starts[b];

    for (int t = tid; t < SEQ_LEN; t += 1024) {
        float v = score[(size_t)b * S + st + t];
        unsigned int bits = __float_as_uint(v);
        unsigned int mono = (bits & 0x80000000u) ? ~bits : (bits | 0x80000000u);
        skey[t] = ~mono;
        sidx[t] = (unsigned short)t;
    }
    __syncthreads();

    for (int k = 2; k <= SEQ_LEN; k <<= 1) {
        for (int j = k >> 1; j > 0; j >>= 1) {
            for (int t = tid; t < SEQ_LEN; t += 1024) {
                int ixj = t ^ j;
                if (ixj > t) {
                    bool up = ((t & k) == 0);
                    unsigned int   k0 = skey[t], k1 = skey[ixj];
                    unsigned short i0 = sidx[t], i1 = sidx[ixj];
                    bool gt = (k0 > k1) || (k0 == k1 && i0 > i1);
                    if (gt == up) {
                        skey[t] = k1; skey[ixj] = k0;
                        sidx[t] = i1; sidx[ixj] = i0;
                    }
                }
            }
            __syncthreads();
        }
    }

    for (int t = tid; t < TOPK; t += 1024) {
        unsigned int mono = ~skey[t];
        unsigned int bits = (mono & 0x80000000u) ? (mono & 0x7fffffffu) : ~mono;
        int idx = (int)sidx[t];
        vals_out[(size_t)b * TOPK + t]  = __uint_as_float(bits);
        pages_out[(size_t)b * TOPK + t] = src_page_table[(size_t)b * S + idx];
    }
}

extern "C" void kernel_launch(void* const* d_in, const int* in_sizes, int n_in,
                              void* d_out, int out_size, void* d_ws, size_t ws_size,
                              hipStream_t stream) {
    const unsigned short* q  = (const unsigned short*)d_in[0];
    const unsigned short* k  = (const unsigned short*)d_in[1];
    const int* spt           = (const int*)d_in[2];
    const int* row_starts    = (const int*)d_in[3];

    const int B = in_sizes[0] / D_DIM;
    const int S = (int)((long long)in_sizes[1] / ((long long)B * D_DIM));

    unsigned char* ws = (unsigned char*)d_ws;
    size_t q8_bytes = ((size_t)B * D_DIM + 255) & ~(size_t)255;
    size_t qs_bytes = ((size_t)B * NGROUP * sizeof(float) + 255) & ~(size_t)255;
    unsigned int* q8w   = (unsigned int*)ws;
    float*        qsw   = (float*)(ws + q8_bytes);
    float*        scorw = (float*)(ws + q8_bytes + qs_bytes);

    int*   pages_out = (int*)d_out;
    float* vals_out  = (float*)d_out + (size_t)B * TOPK;

    quant_q_kernel<<<B, 128, 0, stream>>>(q, q8w, qsw);
    score_kernel<<<B * (S / 128), 256, 0, stream>>>(
        k, (const unsigned char*)q8w, qsw, scorw, S);
    topk_kernel<<<B, 1024, 0, stream>>>(
        scorw, spt, row_starts, pages_out, vals_out, S);
}
